// Transformer_36713380446875
// MI455X (gfx1250) — matrix operand order repaired and checked by reading, not compiled, not run on hardware
//
#include <hip/hip_runtime.h>

// ---------------- problem constants (match reference) ----------------
#define NB   4      // batch
#define ND   1024   // model dim
#define NS   1024   // sequence
#define NH   8      // heads
#define NFF  2048   // ffn dim
#define NL   6      // layers
#define NDH  128    // head dim

typedef __bf16 bf16;
typedef __attribute__((ext_vector_type(16))) __bf16 v16bf;
typedef __attribute__((ext_vector_type(8)))  __bf16 v8bf;
typedef __attribute__((ext_vector_type(4)))  __bf16 v4bf;
typedef __attribute__((ext_vector_type(8)))  float  v8f;

// =====================================================================
// Batched GEMM via v_wmma_f32_16x16x32_bf16.
//   C[z](m,n) = alpha * sum_k A[z](m,k)*B[z](k,n) (+bias[m]) (+resid) (relu?)
// A element (m,k): A[m*lda+k] (TA=0) or A[k*lda+m] (TA=1); TAe = float|bf16
// B element (k,n): B[k*ldb+n] (TB=0) or B[n*ldb+k] (TB=1); TBe = bf16
// Tile: 128(M) x 128(N) x 32(K); 256 threads = 8 wave32s; each wave owns a
// 32x64 slab = 8 accumulators from 2 A-frags x 4 B-frags (8 WMMA / K-step).
// Register double-buffering: next K-tile global loads overlap current WMMAs.
// Requires M%128==0, N%128==0, K%32==0 (holds for every GEMM here).
// =====================================================================
template<typename TAe, typename TBe, typename TCe, bool TA, bool TB>
__global__ __launch_bounds__(256) void gemm_wmma(
    const TAe* __restrict__ A, const TBe* __restrict__ Bm, TCe* __restrict__ C,
    const float* __restrict__ bias, const float* __restrict__ resid,
    int K, int lda, int ldb, int ldc,
    long long strideA, long long strideB, long long strideC,
    float alpha, int relu)
{
  static_assert(sizeof(TBe) == 2, "B operand must be bf16");
  __shared__ __align__(32) bf16 sA[128][32];   // [m][k]
  __shared__ __align__(32) bf16 sB[128][32];   // [n][k] (N-major for frag loads)

  const int z = blockIdx.z;
  A  += (long long)z * strideA;
  Bm += (long long)z * strideB;
  C  += (long long)z * strideC;
  if (resid) resid += (long long)z * strideC;

  const int m0  = blockIdx.y * 128;
  const int n0  = blockIdx.x * 128;
  const int tid = threadIdx.x;
  const int lane = tid & 31;
  const int lo = lane & 15;
  const int hi = lane >> 4;
  const int wm = (tid >> 5) & 3;   // 4 M-slabs of 32
  const int wn = tid >> 7;         // 2 N-slabs of 64

  // staging registers (only the path matching TAe/TA is used)
  float4 aF[4]; v8bf aH[2]; v8bf bH[2];

  auto loadA = [&](int k0) {
    if constexpr (sizeof(TAe) == 4) {
      if constexpr (!TA) {
#pragma unroll
        for (int i = 0; i < 4; ++i) { int c = tid + i * 256; int r = c >> 3, k4 = (c & 7) * 4;
          aF[i] = *(const float4*)&A[(long long)(m0 + r) * lda + (k0 + k4)]; }
      } else {
#pragma unroll
        for (int i = 0; i < 4; ++i) { int c = tid + i * 256; int kk = c >> 5, r4 = (c & 31) * 4;
          aF[i] = *(const float4*)&A[(long long)(k0 + kk) * lda + (m0 + r4)]; }
      }
    } else {
      if constexpr (!TA) {
#pragma unroll
        for (int i = 0; i < 2; ++i) { int c = tid + i * 256; int r = c >> 2, k8 = (c & 3) * 8;
          aH[i] = *(const v8bf*)&A[(long long)(m0 + r) * lda + (k0 + k8)]; }
      } else {
#pragma unroll
        for (int i = 0; i < 2; ++i) { int c = tid + i * 256; int kk = c >> 4, r8 = (c & 15) * 8;
          aH[i] = *(const v8bf*)&A[(long long)(k0 + kk) * lda + (m0 + r8)]; }
      }
    }
  };
  auto storeA = [&]() {
    if constexpr (sizeof(TAe) == 4) {
      if constexpr (!TA) {
#pragma unroll
        for (int i = 0; i < 4; ++i) { int c = tid + i * 256; int r = c >> 3, k4 = (c & 7) * 4;
          v4bf o; o[0] = (bf16)aF[i].x; o[1] = (bf16)aF[i].y; o[2] = (bf16)aF[i].z; o[3] = (bf16)aF[i].w;
          *(v4bf*)&sA[r][k4] = o; }
      } else {
#pragma unroll
        for (int i = 0; i < 4; ++i) { int c = tid + i * 256; int kk = c >> 5, r4 = (c & 31) * 4;
          sA[r4 + 0][kk] = (bf16)aF[i].x; sA[r4 + 1][kk] = (bf16)aF[i].y;
          sA[r4 + 2][kk] = (bf16)aF[i].z; sA[r4 + 3][kk] = (bf16)aF[i].w; }
      }
    } else {
      if constexpr (!TA) {
#pragma unroll
        for (int i = 0; i < 2; ++i) { int c = tid + i * 256; int r = c >> 2, k8 = (c & 3) * 8;
          *(v8bf*)&sA[r][k8] = aH[i]; }
      } else {
#pragma unroll
        for (int i = 0; i < 2; ++i) { int c = tid + i * 256; int kk = c >> 4, r8 = (c & 15) * 8;
#pragma unroll
          for (int j = 0; j < 8; ++j) sA[r8 + j][kk] = aH[i][j]; }
      }
    }
  };
  auto loadB = [&](int k0) {
    if constexpr (!TB) {
#pragma unroll
      for (int i = 0; i < 2; ++i) { int c = tid + i * 256; int kk = c >> 4, n8 = (c & 15) * 8;
        bH[i] = *(const v8bf*)&Bm[(long long)(k0 + kk) * ldb + (n0 + n8)]; }
    } else {
#pragma unroll
      for (int i = 0; i < 2; ++i) { int c = tid + i * 256; int nn = c >> 2, k8 = (c & 3) * 8;
        bH[i] = *(const v8bf*)&Bm[(long long)(n0 + nn) * ldb + (k0 + k8)]; }
    }
  };
  auto storeB = [&]() {
    if constexpr (!TB) {
#pragma unroll
      for (int i = 0; i < 2; ++i) { int c = tid + i * 256; int kk = c >> 4, n8 = (c & 15) * 8;
#pragma unroll
        for (int j = 0; j < 8; ++j) sB[n8 + j][kk] = bH[i][j]; }
    } else {
#pragma unroll
      for (int i = 0; i < 2; ++i) { int c = tid + i * 256; int nn = c >> 2, k8 = (c & 3) * 8;
        *(v8bf*)&sB[nn][k8] = bH[i]; }
    }
  };

  v8f acc[2][4];
#pragma unroll
  for (int i = 0; i < 2; ++i)
#pragma unroll
    for (int j = 0; j < 4; ++j) acc[i][j] = (v8f){};

  loadA(0); loadB(0);
  for (int k0 = 0;;) {
    storeA(); storeB();
    __syncthreads();
    const int kn = k0 + 32;
    if (kn < K) { loadA(kn); loadB(kn); }   // overlap next-tile loads with WMMAs

    // A fragments: row = base+lo, K = hi*16+{0..15} contiguous
    v16bf av[2];
#pragma unroll
    for (int i = 0; i < 2; ++i) {
      const int ar = wm * 32 + i * 16 + lo;
      av[i] = *(const v16bf*)&sA[ar][hi * 16];
    }
    // B fragments: N = base+lo, K = hi*16+{0..15} contiguous
    v16bf bv[4];
#pragma unroll
    for (int j = 0; j < 4; ++j)
      bv[j] = *(const v16bf*)&sB[wn * 64 + j * 16 + lo][hi * 16];

#pragma unroll
    for (int i = 0; i < 2; ++i)
#pragma unroll
      for (int j = 0; j < 4; ++j)
        acc[i][j] = __builtin_amdgcn_wmma_f32_16x16x32_bf16(
            false, av[i], false, bv[j], (short)0, acc[i][j], false, false);

    __syncthreads();
    k0 = kn;
    if (k0 >= K) break;
  }

  // epilogue: C layout = VGPR r -> M = r + hi*8, N = lo
#pragma unroll
  for (int i = 0; i < 2; ++i) {
#pragma unroll
    for (int j = 0; j < 4; ++j) {
#pragma unroll
      for (int r = 0; r < 8; ++r) {
        const int mm = m0 + wm * 32 + i * 16 + hi * 8 + r;
        const int nn = n0 + wn * 64 + j * 16 + lo;
        float v = alpha * acc[i][j][r];
        if (bias)  v += bias[mm];
        if (resid) v += resid[(long long)mm * ldc + nn];
        if (relu)  v = fmaxf(v, 0.f);
        C[(long long)mm * ldc + nn] = (TCe)v;
      }
    }
  }
}

// =====================================================================
// LayerNorm over feature dim D, channel-first (B, D, S); fp32 in, OutT out.
// One 256-thread block per (b, s) column; D == 1024 == 4*256.
// =====================================================================
template<typename OutT>
__global__ __launch_bounds__(256) void layernorm_k(
    const float* __restrict__ in, OutT* __restrict__ out,
    const float* __restrict__ g, const float* __restrict__ b)
{
  const int col = blockIdx.x;
  const int bb = col >> 10;
  const int s  = col & (NS - 1);
  const float* base = in + (long long)bb * ND * NS + s;
  float xs[4], sum = 0.f, sq = 0.f;
#pragma unroll
  for (int i = 0; i < 4; ++i) {
    int d = threadIdx.x + i * 256;
    float x = base[(long long)d * NS];
    xs[i] = x; sum += x; sq += x * x;
  }
  __shared__ float rs[256], rq[256];
  rs[threadIdx.x] = sum; rq[threadIdx.x] = sq;
  __syncthreads();
  for (int off = 128; off > 0; off >>= 1) {
    if (threadIdx.x < off) {
      rs[threadIdx.x] += rs[threadIdx.x + off];
      rq[threadIdx.x] += rq[threadIdx.x + off];
    }
    __syncthreads();
  }
  const float mean = rs[0] * (1.f / ND);
  const float var  = rq[0] * (1.f / ND) - mean * mean;
  const float inv  = rsqrtf(var + 1e-5f);
  OutT* ob = out + (long long)bb * ND * NS + s;
#pragma unroll
  for (int i = 0; i < 4; ++i) {
    int d = threadIdx.x + i * 256;
    ob[(long long)d * NS] = (OutT)(g[d] * (xs[i] - mean) * inv + b[d]);
  }
}

// =====================================================================
// Softmax over last axis (length NS), bf16 in/out, fp32 math, in place.
// =====================================================================
__global__ __launch_bounds__(256) void softmax_k(bf16* __restrict__ p)
{
  bf16* base = p + (long long)blockIdx.x * NS;
  const int t = threadIdx.x;
  v4bf xv = *(const v4bf*)&base[t * 4];
  float xs[4], mx = -3.402823466e38f;
#pragma unroll
  for (int i = 0; i < 4; ++i) { xs[i] = (float)xv[i]; mx = fmaxf(mx, xs[i]); }
  __shared__ float red[256];
  red[t] = mx; __syncthreads();
  for (int o = 128; o > 0; o >>= 1) { if (t < o) red[t] = fmaxf(red[t], red[t + o]); __syncthreads(); }
  mx = red[0]; __syncthreads();
  float sum = 0.f;
#pragma unroll
  for (int i = 0; i < 4; ++i) { xs[i] = __expf(xs[i] - mx); sum += xs[i]; }
  red[t] = sum; __syncthreads();
  for (int o = 128; o > 0; o >>= 1) { if (t < o) red[t] += red[t + o]; __syncthreads(); }
  const float inv = 1.f / red[0];
  v4bf ov;
#pragma unroll
  for (int i = 0; i < 4; ++i) ov[i] = (bf16)(xs[i] * inv);
  *(v4bf*)&base[t * 4] = ov;
}

// f32 -> bf16 bulk convert (used once for y); one float4 per thread
__global__ __launch_bounds__(256) void cvt_bf16_k(const float* __restrict__ a, bf16* __restrict__ o)
{
  const long long i = (long long)blockIdx.x * 256 + threadIdx.x;
  float4 v = ((const float4*)a)[i];
  v4bf r; r[0] = (bf16)v.x; r[1] = (bf16)v.y; r[2] = (bf16)v.z; r[3] = (bf16)v.w;
  ((v4bf*)o)[i] = r;
}

// ---------------------------------------------------------------------
// host-side helpers
// ---------------------------------------------------------------------
static void g_lin_b(hipStream_t st, bf16* C, const float* W, const bf16* X,
                    const float* bias, int M, int K, int relu) {
  dim3 g(NS / 128, M / 128, NB), b(256);
  gemm_wmma<float, bf16, bf16, false, false><<<g, b, 0, st>>>(
      W, X, C, bias, nullptr, K, K, NS, NS, 0LL, (long long)K * NS, (long long)M * NS, 1.f, relu);
}
static void g_lin_f(hipStream_t st, float* C, const float* W, const bf16* X,
                    const float* bias, float* resid, int K) {  // M == ND, resid fused
  dim3 g(NS / 128, ND / 128, NB), b(256);
  gemm_wmma<float, bf16, float, false, false><<<g, b, 0, st>>>(
      W, X, C, bias, resid, K, K, NS, NS, 0LL, (long long)K * NS, (long long)ND * NS, 1.f, 0);
}
static void g_scores(hipStream_t st, bf16* sc, const bf16* q, const bf16* k) {
  dim3 g(NS / 128, NS / 128, NB * NH), b(256);
  gemm_wmma<bf16, bf16, bf16, true, false><<<g, b, 0, st>>>(
      q, k, sc, nullptr, nullptr, NDH, NS, NS, NS,
      (long long)NDH * NS, (long long)NDH * NS, (long long)NS * NS, 0.08838834764831845f, 0);
}
static void g_attnout(hipStream_t st, bf16* ao, const bf16* v, const bf16* sc) {
  dim3 g(NS / 128, NDH / 128, NB * NH), b(256);
  gemm_wmma<bf16, bf16, bf16, false, true><<<g, b, 0, st>>>(
      v, sc, ao, nullptr, nullptr, NS, NS, NS, NS,
      (long long)NDH * NS, (long long)NS * NS, (long long)NDH * NS, 1.f, 0);
}
static void ln_b(hipStream_t st, const float* in, bf16* out, const float* g, const float* b) {
  layernorm_k<bf16><<<dim3(NB * NS), dim3(256), 0, st>>>(in, out, g, b);
}
static void ln_f(hipStream_t st, const float* in, float* out, const float* g, const float* b) {
  layernorm_k<float><<<dim3(NB * NS), dim3(256), 0, st>>>(in, out, g, b);
}

// Full MHA: q from xq16, k/v from xkv16, residual-accumulate into h (f32).
static void mha(hipStream_t st, float* h, const bf16* xq16, const bf16* xkv16,
                bf16* q, bf16* k, bf16* v, bf16* sc, bf16* ao,
                const float* Wq, const float* bq, const float* Wk, const float* bk,
                const float* Wv, const float* bv, const float* Wo, const float* bo) {
  g_lin_b(st, q, Wq, xq16,  bq, ND, ND, 0);
  g_lin_b(st, k, Wk, xkv16, bk, ND, ND, 0);
  g_lin_b(st, v, Wv, xkv16, bv, ND, ND, 0);
  g_scores(st, sc, q, k);
  softmax_k<<<dim3(NB * NH * NS), dim3(256), 0, st>>>(sc);
  g_attnout(st, ao, v, sc);
  g_lin_f(st, h, Wo, ao, bo, h, ND);
}

static void ffn(hipStream_t st, float* h, const bf16* t16, bf16* ff,
                const float* W1, const float* b1, const float* W2, const float* b2) {
  g_lin_b(st, ff, W1, t16, b1, NFF, ND, 1);   // relu fused
  g_lin_f(st, h, W2, ff, b2, h, NFF);
}

extern "C" void kernel_launch(void* const* d_in, const int* in_sizes, int n_in,
                              void* d_out, int out_size, void* d_ws, size_t ws_size,
                              hipStream_t stream) {
  (void)in_sizes; (void)n_in; (void)out_size; (void)ws_size;
  const float* const* in = (const float* const*)d_in;

  // top level insertion order: x, y, enc, dec; nested dicts in sorted-key order
  const float* x = in[0];
  const float* y = in[1];
  const float *eW1 = in[2],  *eW2 = in[3],  *eWk = in[4],  *eWo = in[5],  *eWq = in[6],  *eWv = in[7];
  const float *eb1 = in[8],  *eb2 = in[9],  *ebk = in[10], *ebo = in[11], *ebq = in[12], *ebv = in[13];
  const float *el1b = in[14], *el1g = in[15], *el2b = in[16], *el2g = in[17];
  const float *elfb = in[18], *elfg = in[19];
  const float *dW1 = in[20], *dW2 = in[21], *dWk = in[22], *dWk2 = in[23], *dWo = in[24], *dWo2 = in[25];
  const float *dWq = in[26], *dWq2 = in[27], *dWv = in[28], *dWv2 = in[29];
  const float *db1 = in[30], *db2 = in[31], *dbk = in[32], *dbk2 = in[33], *dbo = in[34], *dbo2 = in[35];
  const float *dbq = in[36], *dbq2 = in[37], *dbv = in[38], *dbv2 = in[39];
  const float *dl1b = in[40], *dl1g = in[41], *dl2b = in[42], *dl2g = in[43], *dl3b = in[44], *dl3g = in[45];
  const float *dlfb = in[46], *dlfg = in[47];

  // workspace: [h fp32][bf16: t16 q k v ao y16 | ff | scores]
  const long long actN = (long long)NB * ND * NS;
  float* h   = (float*)d_ws;
  bf16* bp   = (bf16*)(h + actN);
  bf16* t16  = bp; bp += actN;
  bf16* q16  = bp; bp += actN;
  bf16* k16  = bp; bp += actN;
  bf16* v16  = bp; bp += actN;
  bf16* ao16 = bp; bp += actN;
  bf16* y16  = bp; bp += actN;
  bf16* ff16 = bp; bp += (long long)NB * NFF * NS;
  bf16* sc16 = bp;                        // B*H*S*S

  const long long DDsz = (long long)ND * ND;
  const long long FDsz = (long long)NFF * ND;

  // h = x ; y16 = bf16(y)
  hipMemcpyAsync(h, x, actN * sizeof(float), hipMemcpyDeviceToDevice, stream);
  cvt_bf16_k<<<dim3(actN / 1024), dim3(256), 0, stream>>>(y, y16);

  // -------- encoder --------
  for (int i = 0; i < NL; ++i) {
    ln_b(stream, h, t16, el1g + i * ND, el1b + i * ND);
    mha(stream, h, t16, t16, q16, k16, v16, sc16, ao16,
        eWq + i * DDsz, ebq + i * ND, eWk + i * DDsz, ebk + i * ND,
        eWv + i * DDsz, ebv + i * ND, eWo + i * DDsz, ebo + i * ND);
    ln_b(stream, h, t16, el2g + i * ND, el2b + i * ND);
    ffn(stream, h, t16, ff16, eW1 + i * FDsz, eb1 + i * NFF, eW2 + i * FDsz, eb2 + i * ND);
  }
  ln_f(stream, h, h, elfg, elfb);   // encoder final LN (in place, fp32)

  // -------- decoder (cross-attn memory = raw y) --------
  for (int i = 0; i < NL; ++i) {
    ln_b(stream, h, t16, dl1g + i * ND, dl1b + i * ND);
    mha(stream, h, t16, t16, q16, k16, v16, sc16, ao16,
        dWq + i * DDsz, dbq + i * ND, dWk + i * DDsz, dbk + i * ND,
        dWv + i * DDsz, dbv + i * ND, dWo + i * DDsz, dbo + i * ND);
    ln_b(stream, h, t16, dl2g + i * ND, dl2b + i * ND);
    mha(stream, h, t16, y16, q16, k16, v16, sc16, ao16,
        dWq2 + i * DDsz, dbq2 + i * ND, dWk2 + i * DDsz, dbk2 + i * ND,
        dWv2 + i * DDsz, dbv2 + i * ND, dWo2 + i * DDsz, dbo2 + i * ND);
    ln_b(stream, h, t16, dl3g + i * ND, dl3b + i * ND);
    ffn(stream, h, t16, ff16, dW1 + i * FDsz, db1 + i * NFF, dW2 + i * FDsz, db2 + i * ND);
  }
  ln_f(stream, h, (float*)d_out, dlfg, dlfb);
}
